// GemResNetBlock_13005160972930
// MI455X (gfx1250) — matrix operand order repaired and checked by reading, not compiled, not run on hardware
//
#include <hip/hip_runtime.h>

typedef __attribute__((ext_vector_type(16))) _Float16 v16h;
typedef __attribute__((ext_vector_type(8)))  _Float16 v8h;
typedef __attribute__((ext_vector_type(8)))  float    v8f;

#define WPB 4   // waves per block (wave32)

// ---------------------------------------------------------------------------
// Repack W[o=16][c=16][p=5][q=5][f=5][r=2] (f32) into WMMA-B fragment order,
// f16.  k = (f*2+r)*80 + (c*5+q) in [0,800) = 25 ksteps x 32.  n = o*5+p in
// [0,80) = 5 ntiles x 16.  Fragment layout (matches B VGPR layout for
// v_wmma_f32_16x16x32_f16): flat = ((kk*5+nt)*32 + lane)*16 + h, where lane
// 0-15 hold k-locals 0..7,16..23 (col n = nt*16+lane), lanes 16-31 hold 8..15,24..31.
// ---------------------------------------------------------------------------
__global__ void repack_w(const float* __restrict__ W, _Float16* __restrict__ out) {
  int idx = blockIdx.x * blockDim.x + threadIdx.x;
  if (idx >= 25 * 5 * 32 * 16) return;
  int h    = idx & 15;
  int lane = (idx >> 4) & 31;
  int nt   = (idx >> 9) % 5;
  int kk   = idx / 2560;
  int n    = nt * 16 + (lane & 15);
  int kloc = (h & 8) * 2 + ((lane < 16) ? 0 : 8) + (h & 7);
  int k    = kk * 32 + kloc;
  int fr = k / 80, cq = k % 80;
  int f = fr >> 1, r = fr & 1;
  int c = cq / 5,  q = cq % 5;
  int o = n / 5,   p = n % 5;
  out[idx] = (_Float16)W[((((o * 16 + c) * 5 + p) * 5 + q) * 5 + f) * 2 + r];
}

// Self-interaction Ws[o=16][c=16][p=5][q=5]: k = c*5+q in [0,80), zero-padded
// to 96 (3 ksteps).
__global__ void repack_ws(const float* __restrict__ Ws, _Float16* __restrict__ out) {
  int idx = blockIdx.x * blockDim.x + threadIdx.x;
  if (idx >= 3 * 5 * 32 * 16) return;
  int h    = idx & 15;
  int lane = (idx >> 4) & 31;
  int nt   = (idx >> 9) % 5;
  int kk   = idx / 2560;
  int n    = nt * 16 + (lane & 15);
  int kloc = (h & 8) * 2 + ((lane < 16) ? 0 : 8) + (h & 7);
  int k    = kk * 32 + kloc;
  float v = 0.f;
  if (k < 80) {
    int c = k / 5, q = k % 5;
    int o = n / 5, p = n % 5;
    v = Ws[((o * 16 + c) * 5 + p) * 5 + q];
  }
  out[idx] = (_Float16)v;
}

__global__ void zero_f32(float* __restrict__ p, int n) {
  int i = blockIdx.x * blockDim.x + threadIdx.x;
  if (i < n) p[i] = 0.f;
}

// ---------------------------------------------------------------------------
// Edge message GEMM: one wave per PAIR of 16-edge tiles (32 edges), sharing
// every B fragment between two wmma (M-blocking).
//   A[e, k] = T[e, cq] * P[e, fr]  (k = fr*80+cq)
//   msg = A(16x800) x Wfrag(800x80) via v_wmma_f32_16x16x32_f16
//   scatter: atomicAdd into agg[dst][80].
// Weight fragments (128 KB) live in LDS per workgroup of 4 waves.  Each
// lane's A-side data is register-resident: across all 25 ksteps a lane only
// uses the 5 Th chunks at cq = {0,16,32,48,64} + half*8 and the 10 P values,
// with compile-time indices after full unroll.  All 5 B fragments of a
// k-step are loaded into registers before the 10 wmma consume them, so the
// DS latency is covered by the wmma stream (staggered s_wait_dscnt).
// ---------------------------------------------------------------------------
__global__ __launch_bounds__(128) void edge_conv(
    const float* __restrict__ xin, const long long* __restrict__ eidx,
    const float* __restrict__ precomp, const float* __restrict__ conn,
    const _Float16* __restrict__ Wfrag, float* __restrict__ agg, int nPairs) {
  __shared__ __align__(32) _Float16 sW[25 * 5 * 32 * 16];   // 128000 B
  __shared__ __align__(32) _Float16 sTh[WPB][2][16 * 80];   // transported feats
  __shared__ __align__(32) _Float16 sP[WPB][2][16 * 10];    // precomp, fr-major
  __shared__ int sDst[WPB][2][16];

  const int tid  = threadIdx.x;
  const int lane = tid & 31;
  const int wv   = tid >> 5;

  // cooperative 16B-chunk load of weight fragments into LDS
  {
    const uint4* s4 = (const uint4*)Wfrag;
    uint4* d4 = (uint4*)sW;
    for (int i = tid; i < (25 * 5 * 32 * 16) / 8; i += blockDim.x) d4[i] = s4[i];
  }
  __syncthreads();

  const int eLocal = lane & 15;
  const int half   = lane >> 4;

  for (int pair = blockIdx.x * WPB + wv; pair < nPairs; pair += gridDim.x * WPB) {
#pragma unroll
    for (int u = 0; u < 2; ++u) {
      const int edge = (pair * 2 + u) * 16 + eLocal;
      const int src  = (int)eidx[2 * edge + 0];
      const int dst  = (int)eidx[2 * edge + 1];
      const float phi = conn[edge];
      const float s1 = __sinf(phi), c1 = __cosf(phi);
      const float c2 = c1 * c1 - s1 * s1, s2 = 2.f * s1 * c1;

      // parallel transport: lane pair (eLocal, half) covers 8 channels each
      const float* xr = xin + (size_t)src * 80 + half * 40;
      _Float16* th = &sTh[wv][u][eLocal * 80 + half * 40];
#pragma unroll
      for (int ch = 0; ch < 8; ++ch) {
        float v0 = xr[ch * 5 + 0], a = xr[ch * 5 + 1], b = xr[ch * 5 + 2];
        float d  = xr[ch * 5 + 3], e = xr[ch * 5 + 4];
        th[ch * 5 + 0] = (_Float16)v0;
        th[ch * 5 + 1] = (_Float16)(c1 * a - s1 * b);
        th[ch * 5 + 2] = (_Float16)(s1 * a + c1 * b);
        th[ch * 5 + 3] = (_Float16)(c2 * d - s2 * e);
        th[ch * 5 + 4] = (_Float16)(s2 * d + c2 * e);
      }
      if (half == 0) {
#pragma unroll
        for (int fr = 0; fr < 10; ++fr)
          sP[wv][u][eLocal * 10 + fr] = (_Float16)precomp[edge * 10 + fr];
        sDst[wv][u][eLocal] = dst;
      }
    }
    // per-wave LDS visibility (regions are wave-private; DS ops are in-order)
    asm volatile("s_wait_dscnt 0x0" ::: "memory");

    // Preload both rows' A-side data into registers.
    v8h tv0[5], tv1[5];
    _Float16 pv0[10], pv1[10];
    {
      const _Float16* th0 = &sTh[wv][0][eLocal * 80];
      const _Float16* th1 = &sTh[wv][1][eLocal * 80];
#pragma unroll
      for (int j = 0; j < 5; ++j) {
        tv0[j] = *(const v8h*)(th0 + j * 16 + half * 8);
        tv1[j] = *(const v8h*)(th1 + j * 16 + half * 8);
      }
      const _Float16* p0r = &sP[wv][0][eLocal * 10];
      const _Float16* p1r = &sP[wv][1][eLocal * 10];
#pragma unroll
      for (int i = 0; i < 10; ++i) { pv0[i] = p0r[i]; pv1[i] = p1r[i]; }
    }

    v8f accA[5], accB[5];
#pragma unroll
    for (int nt = 0; nt < 5; ++nt) {
      accA[nt] = (v8f){0,0,0,0,0,0,0,0};
      accB[nt] = (v8f){0,0,0,0,0,0,0,0};
    }

#pragma unroll
    for (int kk = 0; kk < 25; ++kk) {
      // chunk bases; half*8 never crosses an fr (80) boundary, so the fr
      // indices and tv indices below are compile-time constants.
      const int j0 = ((kk * 32) % 80) / 16,      f0 = (kk * 32) / 80;
      const int j1 = ((kk * 32 + 16) % 80) / 16, f1 = (kk * 32 + 16) / 80;
      v16h aA, aB;
#pragma unroll
      for (int i = 0; i < 8; ++i) {
        aA[i]     = tv0[j0][i] * pv0[f0];
        aA[8 + i] = tv0[j1][i] * pv0[f1];
        aB[i]     = tv1[j0][i] * pv1[f0];
        aB[8 + i] = tv1[j1][i] * pv1[f1];
      }
      // batch-load all 5 B fragments of this k-step, then run the 10 wmma
      const _Float16* wk = &sW[kk * 2560 + lane * 16];
      v16h bfr[5];
#pragma unroll
      for (int nt = 0; nt < 5; ++nt) bfr[nt] = *(const v16h*)(wk + nt * 512);
#pragma unroll
      for (int nt = 0; nt < 5; ++nt) {
        accA[nt] = __builtin_amdgcn_wmma_f32_16x16x32_f16(false, aA, false, bfr[nt], (short)0, accA[nt], false, false);
        accB[nt] = __builtin_amdgcn_wmma_f32_16x16x32_f16(false, aB, false, bfr[nt], (short)0, accB[nt], false, false);
      }
    }

    // C/D layout: lane l, vgpr v -> row v+8*(l>=16), col l%16
#pragma unroll
    for (int v = 0; v < 8; ++v) {
      const int d0 = sDst[wv][0][v + half * 8];
      float* ag0 = agg + (size_t)d0 * 80 + eLocal;
      atomicAdd(ag0 + 0,  accA[0][v]);
      atomicAdd(ag0 + 16, accA[1][v]);
      atomicAdd(ag0 + 32, accA[2][v]);
      atomicAdd(ag0 + 48, accA[3][v]);
      atomicAdd(ag0 + 64, accA[4][v]);
      const int d1 = sDst[wv][1][v + half * 8];
      float* ag1 = agg + (size_t)d1 * 80 + eLocal;
      atomicAdd(ag1 + 0,  accB[0][v]);
      atomicAdd(ag1 + 16, accB[1][v]);
      atomicAdd(ag1 + 32, accB[2][v]);
      atomicAdd(ag1 + 48, accB[3][v]);
      atomicAdd(ag1 + 64, accB[4][v]);
    }
  }
}

// ---------------------------------------------------------------------------
// Self-interaction GEMM: agg[n,op] += x[n,cq] * Ws[cq,op]; one wave per
// 16-node tile, K=80 zero-padded to 96 (3 ksteps).  Rows are disjoint, so
// plain read-modify-write (runs after edge_conv on the stream).
// ---------------------------------------------------------------------------
__global__ __launch_bounds__(128) void self_conv(
    const float* __restrict__ xin, const _Float16* __restrict__ Wsfrag,
    float* __restrict__ agg, int nTiles) {
  const int lane = threadIdx.x & 31;
  const int wv   = threadIdx.x >> 5;
  const int half = lane >> 4;
  const int M    = lane & 15;

  for (int tile = blockIdx.x * WPB + wv; tile < nTiles; tile += gridDim.x * WPB) {
    const float* xr = xin + (size_t)(tile * 16 + M) * 80;
    v8f acc0 = {0,0,0,0,0,0,0,0}, acc1 = acc0, acc2 = acc0, acc3 = acc0, acc4 = acc0;
#pragma unroll
    for (int kk = 0; kk < 3; ++kk) {
      const int kb0 = kk * 32 + half * 8;
      const int kb1 = kb0 + 16;
      v16h a;
#pragma unroll
      for (int i = 0; i < 8; ++i) a[i] = (_Float16)xr[kb0 + i];
      if (kk < 2) {
#pragma unroll
        for (int i = 0; i < 8; ++i) a[8 + i] = (_Float16)xr[kb1 + i];
      } else {
#pragma unroll
        for (int i = 0; i < 8; ++i) a[8 + i] = (_Float16)0.f;  // k >= 80 pad
      }
      const _Float16* wk = Wsfrag + kk * 2560 + lane * 16;
      acc0 = __builtin_amdgcn_wmma_f32_16x16x32_f16(false, a, false, *(const v16h*)(wk + 0 * 512), (short)0, acc0, false, false);
      acc1 = __builtin_amdgcn_wmma_f32_16x16x32_f16(false, a, false, *(const v16h*)(wk + 1 * 512), (short)0, acc1, false, false);
      acc2 = __builtin_amdgcn_wmma_f32_16x16x32_f16(false, a, false, *(const v16h*)(wk + 2 * 512), (short)0, acc2, false, false);
      acc3 = __builtin_amdgcn_wmma_f32_16x16x32_f16(false, a, false, *(const v16h*)(wk + 3 * 512), (short)0, acc3, false, false);
      acc4 = __builtin_amdgcn_wmma_f32_16x16x32_f16(false, a, false, *(const v16h*)(wk + 4 * 512), (short)0, acc4, false, false);
    }
#pragma unroll
    for (int v = 0; v < 8; ++v) {
      float* ag = agg + (size_t)(tile * 16 + v + half * 8) * 80 + M;
      ag[0]  += acc0[v];
      ag[16] += acc1[v];
      ag[32] += acc2[v];
      ag[48] += acc3[v];
      ag[64] += acc4[v];
    }
  }
}

// ---------------------------------------------------------------------------
// Per-node epilogue: bias on order-0, optional residual, then the exact
// Fourier->regular(K=7)->ReLU->Fourier nonlinearity.  One thread per (n,c).
// ---------------------------------------------------------------------------
__global__ void node_post(const float* __restrict__ agg, const float* __restrict__ bias,
                          const float* __restrict__ resid, float* __restrict__ out, int total) {
  int idx = blockIdx.x * blockDim.x + threadIdx.x;
  if (idx >= total) return;
  const int node = idx >> 4;
  const int c    = idx & 15;
  const float* a = agg + (size_t)node * 80 + c * 5;
  float y0 = a[0] + bias[c], y1 = a[1], y2 = a[2], y3 = a[3], y4 = a[4];
  if (resid) {
    const float* r = resid + (size_t)node * 80 + c * 5;
    y0 += r[0]; y1 += r[1]; y2 += r[2]; y3 += r[3]; y4 += r[4];
  }
  float o0 = 0.f, o1 = 0.f, o2 = 0.f, o3 = 0.f, o4 = 0.f;
#pragma unroll
  for (int k = 0; k < 7; ++k) {
    const float th = 0.8975979010256552f * (float)k;  // 2*pi/7
    const float sk = __sinf(th), ck = __cosf(th);
    const float c2k = ck * ck - sk * sk, s2k = 2.f * sk * ck;
    float s = y0 + y1 * ck + y2 * sk + y3 * c2k + y4 * s2k;
    s = fmaxf(s, 0.f);
    o0 += s; o1 += s * ck; o2 += s * sk; o3 += s * c2k; o4 += s * s2k;
  }
  float* w = out + (size_t)node * 80 + c * 5;
  w[0] = o0 * (1.f / 7.f);
  w[1] = o1 * (2.f / 7.f);
  w[2] = o2 * (2.f / 7.f);
  w[3] = o3 * (2.f / 7.f);
  w[4] = o4 * (2.f / 7.f);
}

extern "C" void kernel_launch(void* const* d_in, const int* in_sizes, int n_in,
                              void* d_out, int out_size, void* d_ws, size_t ws_size,
                              hipStream_t stream) {
  (void)in_sizes; (void)n_in; (void)out_size; (void)ws_size;
  const float*     x       = (const float*)d_in[0];
  const long long* eix     = (const long long*)d_in[1];  // int64 edge_index
  const float*     precomp = (const float*)d_in[2];
  const float*     conn    = (const float*)d_in[3];
  const float*     W1      = (const float*)d_in[4];
  const float*     b1      = (const float*)d_in[5];
  const float*     Ws1     = (const float*)d_in[6];
  const float*     W2      = (const float*)d_in[7];
  const float*     b2      = (const float*)d_in[8];
  const float*     Ws2     = (const float*)d_in[9];
  float* out = (float*)d_out;

  char* ws = (char*)d_ws;
  _Float16* W1f  = (_Float16*)(ws + 0);       // 128000 B
  _Float16* W2f  = (_Float16*)(ws + 128000);  // 128000 B
  _Float16* Ws1f = (_Float16*)(ws + 256000);  //  15360 B
  _Float16* Ws2f = (_Float16*)(ws + 271360);  //  15360 B
  float* agg1 = (float*)(ws + 286720);        // 3.2 MB
  float* agg2 = agg1 + 800000;                // 3.2 MB
  float* h    = agg2 + 800000;                // 3.2 MB

  repack_w <<<250, 256, 0, stream>>>(W1, W1f);
  repack_w <<<250, 256, 0, stream>>>(W2, W2f);
  repack_ws<<<30, 256, 0, stream>>>(Ws1, Ws1f);
  repack_ws<<<30, 256, 0, stream>>>(Ws2, Ws2f);
  zero_f32 <<<(1600000 + 255) / 256, 256, 0, stream>>>(agg1, 1600000);

  // layer 1
  edge_conv<<<256, 128, 0, stream>>>(x, eix, precomp, conn, W1f, agg1, 5000);
  self_conv<<<40, 128, 0, stream>>>(x, Ws1f, agg1, 625);
  node_post<<<625, 256, 0, stream>>>(agg1, b1, nullptr, h, 160000);
  // layer 2 (+ residual) -> final nonlin -> out
  edge_conv<<<256, 128, 0, stream>>>(h, eix, precomp, conn, W2f, agg2, 5000);
  self_conv<<<40, 128, 0, stream>>>(h, Ws2f, agg2, 625);
  node_post<<<625, 256, 0, stream>>>(agg2, b2, x, out, 160000);
}
